// Correlation_65180423684747
// MI455X (gfx1250) — matrix operand order repaired and checked by reading, not compiled, not run on hardware
//
#include <hip/hip_runtime.h>

typedef _Float16 v16h __attribute__((ext_vector_type(16)));
typedef _Float16 v8h  __attribute__((ext_vector_type(8)));
typedef _Float16 v4h  __attribute__((ext_vector_type(4)));
typedef float    v8f  __attribute__((ext_vector_type(8)));

#define C_    256
#define H_    64
#define W_    128
#define B_    8
#define DR_   10            // max displacement / stride2
#define DS_   21            // 2*DR_+1
#define KPAD  264           // 256 halves + 8 pad (bank-conflict avoidance)
#define SPAD  136           // 128 floats + 8 pad
#define CHW   (C_ * H_ * W_)
#define HW    (H_ * W_)

// Convert one (C=256, W=128) fp32 row (channel stride HW) to f16 LDS in [x][k]
// layout with padded k-stride KPAD. 256 threads: tid&127 = x, tid>>7 picks a
// 4-channel subgroup; 32 iterations cover all 256 channels; each iteration does
// 4 coalesced global b32 loads and one 8-byte ds_store (4 contiguous k).
__device__ __forceinline__ void load_row_f16(const float* __restrict__ src,
                                             _Float16* __restrict__ dst,
                                             const int tid) {
  const int x  = tid & 127;
  const int cg = (tid >> 7) & 1;
  #pragma unroll 4
  for (int i = 0; i < 32; ++i) {
    const int c = i * 8 + cg * 4;
    const float* p = src + (size_t)c * HW + x;
    const float f0 = p[0];
    const float f1 = p[HW];
    const float f2 = p[2 * HW];
    const float f3 = p[3 * HW];
    v4h h;
    h[0] = (_Float16)f0;
    h[1] = (_Float16)f1;
    h[2] = (_Float16)f2;
    h[3] = (_Float16)f3;
    *(v4h*)(dst + x * KPAD + c) = h;
  }
}

__global__ __launch_bounds__(256, 1)
void Correlation_65180423684747_kernel(const float* __restrict__ in1,
                                       const float* __restrict__ in2,
                                       float* __restrict__ out) {
  __shared__ _Float16 sA[W_ * KPAD];        // in1 row, f16, [x][k]
  __shared__ _Float16 sB[W_ * KPAD];        // in2 row, f16, [x][k]
  __shared__ float    sStage[DS_ * SPAD];   // [di][x] staging for coalesced out

  const int tid  = threadIdx.x;
  const int wid  = tid >> 5;     // wave id == output tile row ti (0..7)
  const int lane = tid & 31;
  const int ln   = lane & 15;
  const int lh   = lane >> 4;    // half-wave select

  const int b = blockIdx.x >> 6; // blocks = B*H
  const int y = blockIdx.x & 63;

  const float* img1 = in1 + (size_t)b * CHW;
  const float* img2 = in2 + (size_t)b * CHW;

  load_row_f16(img1 + (size_t)y * W_, sA, tid);

  // A fragment base: 16-bit A 16x32 layout — lanes 0-15: M=lane, K {0..7,16..23};
  // lanes 16-31: M=lane-16, K {8..15,24..31}. [x][k] layout => contiguous loads.
  const _Float16* aBase = sA + (wid * 16 + ln) * KPAD + lh * 8;

  // B fragment bases for the 5 band tiles, tj CLAMPED to [0,7] so the WMMA
  // loop is completely branch-free (clamped duplicates are rejected in the
  // scatter phase by the x2-range predicate). 16-bit B 32x16 layout:
  // lanes 0-15: N=lane, K {0..7,16..23}; lanes 16-31: N=lane-16, K {8..15,24..31}.
  const _Float16* bBase[5];
  #pragma unroll
  for (int t = 0; t < 5; ++t) {
    int tj = wid - 2 + t;
    tj = (tj < 0) ? 0 : ((tj > 7) ? 7 : tj);
    bBase[t] = sB + (tj * 16 + ln) * KPAD + lh * 8;
  }

  __syncthreads();

  for (int dj = -DR_; dj <= DR_; ++dj) {
    // Zero staging: covers zero-fill for out-of-range x2 and invalid rows.
    for (int i = tid; i < DS_ * SPAD; i += 256) sStage[i] = 0.0f;

    const int  y2       = y + 2 * dj;
    const bool rowValid = (y2 >= 0) && (y2 < H_);
    if (rowValid) load_row_f16(img2 + (size_t)y2 * W_, sB, tid);

    // Hint-prefetch next dj's in2 row into cache (global_prefetch_b8).
    const int y2n = y2 + 2;
    if (dj < DR_ && y2n >= 0 && y2n < H_) {
      const float* pf = img2 + (size_t)y2n * W_ +
                        (size_t)(((tid >> 7) & 1) * 4) * HW + (tid & 127);
      __builtin_prefetch(pf, 0, 1);
    }

    __syncthreads();

    if (rowValid) {
      // Banded Gram: wave wid computes G(ti=wid, tj) for 5 clamped tj tiles.
      v8f acc[5];
      #pragma unroll
      for (int t = 0; t < 5; ++t) acc[t] = (v8f){};

      #pragma unroll
      for (int kk = 0; kk < 8; ++kk) {          // K = 8 chunks of 32
        const v8h a0 = *(const v8h*)(aBase + kk * 32);
        const v8h a1 = *(const v8h*)(aBase + kk * 32 + 16);
        const v16h a = __builtin_shufflevector(
            a0, a1, 0, 1, 2, 3, 4, 5, 6, 7, 8, 9, 10, 11, 12, 13, 14, 15);
        #pragma unroll
        for (int t = 0; t < 5; ++t) {           // branch-free: EXEC stays full
          const v8h b0 = *(const v8h*)(bBase[t] + kk * 32);
          const v8h b1 = *(const v8h*)(bBase[t] + kk * 32 + 16);
          const v16h bb = __builtin_shufflevector(
              b0, b1, 0, 1, 2, 3, 4, 5, 6, 7, 8, 9, 10, 11, 12, 13, 14, 15);
          acc[t] = __builtin_amdgcn_wmma_f32_16x16x32_f16(
              false, a, false, bb, (short)0, acc[t], false, false);
        }
      }

      // Scatter valid even diagonals |x2-x1|<=20 into staging (÷C).
      // D layout: lanes 0-15: N=lane, M=r ; lanes 16-31: N=lane-16, M=r+8.
      // x2o is the UNCLAMPED column: out-of-range x2o rejects the garbage
      // tiles computed with clamped tj (per-lane EXEC masking is fine here).
      #pragma unroll
      for (int t = 0; t < 5; ++t) {
        const int x2o = (wid - 2 + t) * 16 + ln;
        #pragma unroll
        for (int r = 0; r < 8; ++r) {
          const int x1    = wid * 16 + r + lh * 8;
          const int delta = x2o - x1;
          if (x2o >= 0 && x2o < W_ && ((delta & 1) == 0) &&
              (delta >= -2 * DR_) && (delta <= 2 * DR_)) {
            sStage[((delta >> 1) + DR_) * SPAD + x1] = acc[t][r] * (1.0f / C_);
          }
        }
      }
    }

    __syncthreads();

    // Coalesced write of 21 channels x 128 pixels for this dj.
    for (int i = tid; i < DS_ * W_; i += 256) {
      const int di = i >> 7;
      const int xx = i & 127;
      const int ch = (dj + DR_) * DS_ + di;
      out[(((size_t)b * (DS_ * DS_) + ch) * H_ + y) * W_ + xx] =
          sStage[di * SPAD + xx];
    }

    __syncthreads();  // protect sB / sStage reuse next iteration
  }
}

extern "C" void kernel_launch(void* const* d_in, const int* in_sizes, int n_in,
                              void* d_out, int out_size, void* d_ws, size_t ws_size,
                              hipStream_t stream) {
  (void)in_sizes; (void)n_in; (void)d_ws; (void)ws_size; (void)out_size;
  const float* in1 = (const float*)d_in[0];
  const float* in2 = (const float*)d_in[1];
  float*       out = (float*)d_out;
  Correlation_65180423684747_kernel<<<dim3(B_ * H_), dim3(256), 0, stream>>>(
      in1, in2, out);
}
